// FineDeformableTransformerEncoderLayer_73873437491494
// MI455X (gfx1250) — matrix operand order repaired and checked by reading, not compiled, not run on hardware
//
#include <hip/hip_runtime.h>

// ---------------------------------------------------------------------------
// FineDeformableTransformerEncoderLayer for MI455X (gfx1250, wave32, WMMA)
// ---------------------------------------------------------------------------
typedef _Float16 v16h __attribute__((ext_vector_type(16)));
typedef _Float16 v8h  __attribute__((ext_vector_type(8)));
typedef _Float16 h4v  __attribute__((ext_vector_type(4)));
typedef float    v8f  __attribute__((ext_vector_type(8)));
typedef int      i2v  __attribute__((vector_size(2 * sizeof(int))));

#define DM      256
#define DFFN    1024
#define LSP     5440
#define NVT     8
#define LQ      5448          // LSP + NVT
#define NB      4
#define RTOT    (NB * LQ)     // 21792 rows, divisible by 32
#define OUT0    (NB * LSP * DM)            // 5,570,560 floats
#define OUT1    (NB * 600 * DM)            //   614,400 floats (all_vt)
#define OUT01   (OUT0 + OUT1)              // 6,184,960

#define AS1 __attribute__((address_space(1)))
#define AS3 __attribute__((address_space(3)))

#if defined(__AMDGCN__) && \
    __has_builtin(__builtin_amdgcn_global_load_async_to_lds_b64) && \
    __has_builtin(__builtin_amdgcn_s_wait_asynccnt)
#define HAVE_ASYNC_LDS 1
#else
#define HAVE_ASYNC_LDS 0
#endif

static __device__ __forceinline__ float wave_red(float s) {
#pragma unroll
    for (int m = 16; m; m >>= 1) s += __shfl_xor(s, m, 32);
    return s;
}

// ---------------- fp32 -> f16 convert ----------------
__global__ void k_cvt_h(const float* __restrict__ in, _Float16* __restrict__ out, int n) {
    int i = blockIdx.x * blockDim.x + threadIdx.x;
    if (i < n) out[i] = (_Float16)in[i];
}

// ---------------- build q = (src+pos | sel_vt), v_in = (src | sel_vt) -------
__global__ void k_build_qv(const float* __restrict__ src, const float* __restrict__ pos,
                           const float* __restrict__ selvt,
                           _Float16* __restrict__ qh, _Float16* __restrict__ vinh,
                           float* __restrict__ vinf) {
    long idx = (long)blockIdx.x * blockDim.x + threadIdx.x;
    if (idx >= (long)RTOT * DM) return;
    long row = idx >> 8;
    int  col = (int)(idx & 255);
    int  n   = (int)(row / LQ);
    int  q   = (int)(row % LQ);
    float qv, vv;
    if (q < LSP) {
        long o = ((long)n * LSP + q) * DM + col;
        float s = src[o];
        qv = s + pos[o];
        vv = s;
    } else {
        long o = ((long)n * NVT + (q - LSP)) * DM + col;
        float s = selvt[o];
        qv = s; vv = s;
    }
    qh[idx]   = (_Float16)qv;
    vinh[idx] = (_Float16)vv;
    vinf[idx] = vv;
}

// ---------------- WMMA GEMM: C[M,N] = A[M,K](f16) * W[N,K](f16)^T + bias ----
// Block tile 32x128: 8 waves = 2 (M) x 4 (N); each wave holds one A fragment
// and two 16x16 accumulators (cols n0, n0+16). A staged via LDS; on gfx1250
// the staging uses GLOBAL_LOAD_ASYNC_TO_LDS (ASYNCcnt) instead of a VGPR trip.
template <bool RELU, bool MASKED, bool OF, bool OH>
__global__ void k_gemm_wmma(const _Float16* __restrict__ A,
                            const _Float16* __restrict__ W,
                            const float* __restrict__ bias,
                            float* __restrict__ outF, _Float16* __restrict__ outH,
                            const unsigned char* __restrict__ mask,
                            int Nout, int K) {
    __shared__ __align__(32) _Float16 As[32 * 32];
    h4v*       As4 = reinterpret_cast<h4v*>(As);
    const h4v* A4  = reinterpret_cast<const h4v*>(A);

    const int tid  = threadIdx.x;
    const int wave = tid >> 5;
    const int lane = tid & 31;
    const int rc   = lane & 15;     // A row / C col (local)
    const int h    = lane >> 4;     // lane half
    const int mw   = wave >> 2;     // M sub-tile (0..1)
    const int nw   = wave & 3;      // N sub-tile (0..3)
    const int row0 = blockIdx.x * 32;
    const int n0   = blockIdx.y * 128 + nw * 32;

    // staging: 256 threads x one 4xf16 (b64) = 32x32 tile
    const int sidx = tid * 4;
    const int sr   = sidx >> 5;
    const int sk   = sidx & 31;

    v8f acc0 = {}, acc1 = {};
    for (int k0 = 0; k0 < K; k0 += 32) {
        __syncthreads();
#if HAVE_ASYNC_LDS
        __builtin_amdgcn_global_load_async_to_lds_b64(
            (AS1 i2v*)(uintptr_t)(A + ((long)(row0 + sr) * K + k0 + sk)),
            (AS3 i2v*)(unsigned)(uintptr_t)(&As[sidx]), 0, 0);
        __builtin_amdgcn_s_wait_asynccnt(0);
#else
        As4[sidx >> 2] = A4[((long)(row0 + sr) * K + k0 + sk) >> 2];
#endif
        __syncthreads();

        // A fragment: two contiguous 8xf16 runs -> 2x ds_load_b128
        const _Float16* ap = &As[(mw * 16 + rc) * 32 + h * 16];
        v8h alo = *reinterpret_cast<const v8h*>(ap);
        v8h ahi = *reinterpret_cast<const v8h*>(ap + 8);
        v16h va = __builtin_shufflevector(alo, ahi,
                                          0, 1, 2, 3, 4, 5, 6, 7,
                                          8, 9, 10, 11, 12, 13, 14, 15);
        // B fragments: per lane 16 contiguous f16 (K h*16..h*16+15) -> b128 x2
        const _Float16* wp = W + (long)(n0 + rc) * K + k0 + h * 16;
        v16h vb0 = *reinterpret_cast<const v16h*>(wp);
        v16h vb1 = *reinterpret_cast<const v16h*>(wp + (long)16 * K);

        acc0 = __builtin_amdgcn_wmma_f32_16x16x32_f16(false, va, false, vb0,
                                                      (short)0, acc0, false, false);
        acc1 = __builtin_amdgcn_wmma_f32_16x16x32_f16(false, va, false, vb1,
                                                      (short)0, acc1, false, false);
    }

    const int   col0 = n0 + rc, col1 = col0 + 16;
    const float bc0  = bias[col0];
    const float bc1  = bias[col1];
#pragma unroll
    for (int v = 0; v < 8; ++v) {
        const int row = row0 + mw * 16 + v + 8 * h;
        float v0 = acc0[v] + bc0;
        float v1 = acc1[v] + bc1;
        if (RELU) { v0 = fmaxf(v0, 0.0f); v1 = fmaxf(v1, 0.0f); }
        if (MASKED) {
            if (mask[row]) { v0 = 0.0f; v1 = 0.0f; }
        }
        const long o = (long)row * Nout;
        if (OF) { outF[o + col0] = v0; outF[o + col1] = v1; }
        if (OH) { outH[o + col0] = (_Float16)v0; outH[o + col1] = (_Float16)v1; }
    }
}

// ---------------- softmax over 16 (level*point) per head --------------------
__global__ void k_softmax16(float* __restrict__ attn) {
    long g = (long)blockIdx.x * blockDim.x + threadIdx.x;
    if (g >= (long)RTOT * 8) return;
    long  base = (g >> 3) * 128 + (g & 7) * 16;
    float mx = -1e30f;
#pragma unroll
    for (int i = 0; i < 16; ++i) mx = fmaxf(mx, attn[base + i]);
    float e[16], s = 0.0f;
#pragma unroll
    for (int i = 0; i < 16; ++i) { e[i] = __expf(attn[base + i] - mx); s += e[i]; }
    float r = 1.0f / s;
#pragma unroll
    for (int i = 0; i < 16; ++i) attn[base + i] = e[i] * r;
}

// ---------------- multi-scale deformable sampling ---------------------------
// one wave per (n, q, head); lane = channel (Dh = 32 = wave32)
__global__ void k_sample(const float* __restrict__ value, const float* __restrict__ offs,
                         const float* __restrict__ attn, const float* __restrict__ refp,
                         _Float16* __restrict__ out) {
    const int lane = threadIdx.x & 31;
    long gw = (long)blockIdx.x * 8 + (threadIdx.x >> 5);   // [0, RTOT*8)
    const int m  = (int)(gw & 7);
    long t = gw >> 3;
    const int qi = (int)(t % LQ);
    const int n  = (int)(t / LQ);
    const long row = (long)n * LQ + qi;

    const int Hs[4] = {64, 32, 16, 8};
    const int Ws[4] = {64, 32, 16, 8};
    const int St[4] = {0, 4096, 5120, 5376};

    const float* orow = offs + row * 256;
    const float* arow = attn + row * 128;
    float acc = 0.0f;
#pragma unroll
    for (int l = 0; l < 4; ++l) {
        const int   H = Hs[l], Wd = Ws[l], st = St[l];
        const float rx = refp[(row * 4 + l) * 2 + 0];
        const float ry = refp[(row * 4 + l) * 2 + 1];
#pragma unroll
        for (int p = 0; p < 4; ++p) {
            const int oi = ((m * 4 + l) * 4 + p) * 2;
            const float x = (rx + orow[oi]     / (float)Wd) * (float)Wd - 0.5f;
            const float y = (ry + orow[oi + 1] / (float)H)  * (float)H  - 0.5f;
            const float a = arow[m * 16 + l * 4 + p];
            const float x0 = floorf(x), y0 = floorf(y);
            const float wx = x - x0,   wy = y - y0;
            float s = 0.0f;
#pragma unroll
            for (int cy = 0; cy <= 1; ++cy) {
#pragma unroll
                for (int cx = 0; cx <= 1; ++cx) {
                    const float fx = x0 + cx, fy = y0 + cy;
                    float wgt = (cx ? wx : 1.0f - wx) * (cy ? wy : 1.0f - wy);
                    const bool valid = (fx >= 0.0f) && (fx < (float)Wd) &&
                                       (fy >= 0.0f) && (fy < (float)H);
                    int xi = (int)fx; xi = xi < 0 ? 0 : (xi > Wd - 1 ? Wd - 1 : xi);
                    int yi = (int)fy; yi = yi < 0 ? 0 : (yi > H  - 1 ? H  - 1 : yi);
                    const long vrow = (long)n * LQ + st + yi * Wd + xi;
                    const float g = value[vrow * DM + m * 32 + lane];
                    s += (valid ? wgt : 0.0f) * g;
                }
            }
            acc += a * s;
        }
    }
    out[row * DM + m * 32 + lane] = (_Float16)acc;
}

// ---------------- LayerNorm of (a + b); wave per row ------------------------
// SCATTER -> write into d_out tuple layout (LN2), else write f32 + f16
template <bool SCATTER>
__global__ void k_ln(const float* __restrict__ Aa, const float* __restrict__ Bb,
                     const float* __restrict__ g, const float* __restrict__ beta,
                     float* __restrict__ outF, _Float16* __restrict__ outH,
                     float* __restrict__ outScat) {
    const int lane = threadIdx.x & 31;
    const long row = (long)blockIdx.x * 8 + (threadIdx.x >> 5);
    const float* pa = Aa + row * DM;
    const float* pb = Bb + row * DM;
    float v[8], s = 0.0f;
#pragma unroll
    for (int j = 0; j < 8; ++j) {
        int col = lane + 32 * j;
        v[j] = pa[col] + pb[col];
        s += v[j];
    }
    s = wave_red(s);
    const float mean = s * (1.0f / DM);
    float sq = 0.0f;
#pragma unroll
    for (int j = 0; j < 8; ++j) { float d = v[j] - mean; sq += d * d; }
    sq = wave_red(sq);
    const float rs = rsqrtf(sq * (1.0f / DM) + 1e-5f);

    float* dst = nullptr;
    if (SCATTER) {
        int n = (int)(row / LQ), q = (int)(row % LQ);
        dst = (q < LSP) ? outScat + ((long)n * LSP + q) * DM
                        : outScat + OUT01 + ((long)n * NVT + (q - LSP)) * DM;
    }
#pragma unroll
    for (int j = 0; j < 8; ++j) {
        int col = lane + 32 * j;
        float o = g[col] * (v[j] - mean) * rs + beta[col];
        if (SCATTER) dst[col] = o;
        else { outF[row * DM + col] = o; outH[row * DM + col] = (_Float16)o; }
    }
}

// ---------------- passthrough copy (all_vt) ---------------------------------
__global__ void k_copy(const float* __restrict__ in, float* __restrict__ out, int n) {
    int i = blockIdx.x * blockDim.x + threadIdx.x;
    if (i < n) out[i] = in[i];
}

// ---------------------------------------------------------------------------
extern "C" void kernel_launch(void* const* d_in, const int* in_sizes, int n_in,
                              void* d_out, int out_size, void* d_ws, size_t ws_size,
                              hipStream_t stream) {
    const float* src   = (const float*)d_in[0];
    const float* pos   = (const float*)d_in[1];
    const float* refp  = (const float*)d_in[2];
    const unsigned char* pad = (const unsigned char*)d_in[5];
    const float* allvt = (const float*)d_in[6];
    const float* selvt = (const float*)d_in[7];
    const float* Wv = (const float*)d_in[8];  const float* bv  = (const float*)d_in[9];
    const float* Wo = (const float*)d_in[10]; const float* bo  = (const float*)d_in[11];
    const float* Wa = (const float*)d_in[12]; const float* ba  = (const float*)d_in[13];
    const float* Wp = (const float*)d_in[14]; const float* bp  = (const float*)d_in[15];
    const float* g1 = (const float*)d_in[16]; const float* be1 = (const float*)d_in[17];
    const float* W1 = (const float*)d_in[18]; const float* bf1 = (const float*)d_in[19];
    const float* W2 = (const float*)d_in[20]; const float* bf2 = (const float*)d_in[21];
    const float* g2 = (const float*)d_in[22]; const float* be2 = (const float*)d_in[23];

    char* ws = (char*)d_ws;
    size_t off = 0;
    auto take = [&](size_t bytes) -> char* {
        off = (off + 255) & ~(size_t)255;
        char* p = ws + off;
        off += bytes;
        return p;
    };
    const size_t R = RTOT;
    _Float16* q_h    = (_Float16*)take(R * DM * 2);
    _Float16* vin_h  = (_Float16*)take(R * DM * 2);
    float*    vin_f  = (float*)   take(R * DM * 4);
    float*    val_f  = (float*)   take(R * DM * 4);
    float*    off_f  = (float*)   take(R * DM * 4);
    float*    attn_f = (float*)   take(R * 128 * 4);
    _Float16* samp_h = (_Float16*)take(R * DM * 2);
    float*    src2_f = (float*)   take(R * DM * 4);
    float*    x_f    = (float*)   take(R * DM * 4);
    _Float16* x_h    = (_Float16*)take(R * DM * 2);
    _Float16* hh     = (_Float16*)take(R * DFFN * 2);
    float*    h2_f   = (float*)   take(R * DM * 4);
    _Float16* Wv_h = (_Float16*)take(DM * DM * 2);
    _Float16* Wo_h = (_Float16*)take(DM * DM * 2);
    _Float16* Wa_h = (_Float16*)take(128 * DM * 2);
    _Float16* Wp_h = (_Float16*)take(DM * DM * 2);
    _Float16* W1_h = (_Float16*)take(DFFN * DM * 2);
    _Float16* W2_h = (_Float16*)take(DM * DFFN * 2);
    float* out = (float*)d_out;

    // --- weight conversion fp32 -> f16 ---
    auto cvt = [&](const float* w, _Float16* wh, int n) {
        k_cvt_h<<<(n + 255) / 256, 256, 0, stream>>>(w, wh, n);
    };
    cvt(Wv, Wv_h, DM * DM);
    cvt(Wo, Wo_h, DM * DM);
    cvt(Wa, Wa_h, 128 * DM);
    cvt(Wp, Wp_h, DM * DM);
    cvt(W1, W1_h, DFFN * DM);
    cvt(W2, W2_h, DM * DFFN);

    // --- build q / v_in ---
    k_build_qv<<<(int)((R * DM + 255) / 256), 256, 0, stream>>>(src, pos, selvt,
                                                               q_h, vin_h, vin_f);

    // --- projections (WMMA GEMMs); block tile 32x128 ---
    dim3 blk(256);
    const int MB = RTOT / 32;   // 681
    // value = v_in @ Wv^T + bv, then pad-mask zeroing
    k_gemm_wmma<false, true, true, false><<<dim3(MB, DM / 128), blk, 0, stream>>>(
        vin_h, Wv_h, bv, val_f, nullptr, pad, DM, DM);
    // sampling offsets = q @ Wo^T + bo
    k_gemm_wmma<false, false, true, false><<<dim3(MB, DM / 128), blk, 0, stream>>>(
        q_h, Wo_h, bo, off_f, nullptr, nullptr, DM, DM);
    // attention logits = q @ Wa^T + ba
    k_gemm_wmma<false, false, true, false><<<dim3(MB, 1), blk, 0, stream>>>(
        q_h, Wa_h, ba, attn_f, nullptr, nullptr, 128, DM);

    // --- softmax over 16 per head ---
    k_softmax16<<<(int)((R * 8 + 255) / 256), 256, 0, stream>>>(attn_f);

    // --- deformable bilinear sampling ---
    k_sample<<<RTOT, 256, 0, stream>>>(val_f, off_f, attn_f, refp, samp_h);

    // --- output projection src2 = samp @ Wp^T + bp ---
    k_gemm_wmma<false, false, true, false><<<dim3(MB, DM / 128), blk, 0, stream>>>(
        samp_h, Wp_h, bp, src2_f, nullptr, nullptr, DM, DM);

    // --- LN1: x = LN(v_in + src2) ---
    k_ln<false><<<RTOT / 8, 256, 0, stream>>>(vin_f, src2_f, g1, be1, x_f, x_h, nullptr);

    // --- FFN up (+ReLU): h = relu(x @ W1^T + bf1), f16 out ---
    k_gemm_wmma<true, false, false, true><<<dim3(MB, DFFN / 128), blk, 0, stream>>>(
        x_h, W1_h, bf1, nullptr, hh, nullptr, DFFN, DM);
    // --- FFN down: h2 = h @ W2^T + bf2 ---
    k_gemm_wmma<false, false, true, false><<<dim3(MB, DM / 128), blk, 0, stream>>>(
        hh, W2_h, bf2, h2_f, nullptr, nullptr, DM, DFFN);

    // --- LN2: out = LN(x + h2), scattered into tuple layout of d_out ---
    k_ln<true><<<RTOT / 8, 256, 0, stream>>>(x_f, h2_f, g2, be2, nullptr, nullptr, out);

    // --- pass through all_vt (tuple output #2) ---
    k_copy<<<(OUT1 + 255) / 256, 256, 0, stream>>>(allvt, out + OUT0, OUT1);
}
